// WindowAttention_29978871726215
// MI455X (gfx1250) — matrix operand order repaired and checked by reading, not compiled, not run on hardware
//
#include <hip/hip_runtime.h>
#include <hip/hip_bf16.h>

// ---------------------------------------------------------------------------
// Types for WMMA fragments (wave32, CDNA5)
// ---------------------------------------------------------------------------
typedef __attribute__((ext_vector_type(16))) __bf16 bf16x16;
typedef __attribute__((ext_vector_type(8)))  float  v8f;
typedef unsigned int u32x4 __attribute__((ext_vector_type(4)));

union FragB16 { bf16x16 v; u32x4 u[2]; };

#ifndef __has_builtin
#define __has_builtin(x) 0
#endif
#if __has_builtin(__builtin_amdgcn_global_load_async_to_lds_b128)
#define HAS_ASYNC_LDS 1
#else
#define HAS_ASYNC_LDS 0
#endif

// Exact pointee type of the async-LDS builtins: GCC-vector int4 with
// explicit address spaces (global = 1, LDS = 3).
typedef int v4i_gcc __attribute__((vector_size(16)));
typedef __attribute__((address_space(1))) v4i_gcc* gas_v4i_ptr;
typedef __attribute__((address_space(3))) v4i_gcc* las_v4i_ptr;

// Async global->LDS 16-byte copy (CDNA5 GLOBAL_LOAD_ASYNC_TO_LDS_B128,
// tracked by ASYNCcnt).  Falls back to a synchronous copy if the builtin is
// not available on this toolchain.
__device__ __forceinline__ void async_copy_b128(const unsigned short* g,
                                                unsigned short* l) {
#if HAS_ASYNC_LDS
    __builtin_amdgcn_global_load_async_to_lds_b128(
        (gas_v4i_ptr)g, (las_v4i_ptr)l, 0, 0);
#else
    *(u32x4*)l = *(const u32x4*)g;
#endif
}

__device__ __forceinline__ void wait_async() {
#if HAS_ASYNC_LDS
    asm volatile("s_wait_asynccnt 0x0" ::: "memory");
#endif
}

__device__ __forceinline__ unsigned short f2bf(float f) {
    unsigned int u = __float_as_uint(f);
    u += 0x7FFFu + ((u >> 16) & 1u);          // round-to-nearest-even
    return (unsigned short)(u >> 16);
}

__device__ __forceinline__ v8f vzero8() {
    v8f z;
#pragma unroll
    for (int i = 0; i < 8; ++i) z[i] = 0.0f;
    return z;
}

#define WMMA_BF16(A, B, C) \
    __builtin_amdgcn_wmma_f32_16x16x32_bf16(false, (A), false, (B), (short)0, (C), false, false)

// ---------------------------------------------------------------------------
// Problem constants
// ---------------------------------------------------------------------------
#define DIMC     256
#define HEADS    8
#define NTOK     64
#define NW       1024
#define BWIN     4096                 // BATCH*NW windows
#define HDIM     32
#define SCALE_Q  0.17677669529663687f // 32^-0.5

// ---------------------------------------------------------------------------
// Kernel 0: convert weights fp32 -> bf16.  wqkv = [w_q ; w_kv] rows [768][256]
// ---------------------------------------------------------------------------
__global__ void cvt_weights(const float* __restrict__ wq,
                            const float* __restrict__ wkv,
                            const float* __restrict__ wo,
                            unsigned short* __restrict__ wqkv_bf,
                            unsigned short* __restrict__ wo_bf) {
    int i = blockIdx.x * blockDim.x + threadIdx.x;
    if (i < 196608) {
        float v = (i < 65536) ? wq[i] : wkv[i - 65536];
        wqkv_bf[i] = f2bf(v);
    } else if (i < 262144) {
        int j = i - 196608;
        wo_bf[j] = f2bf(wo[j]);
    }
}

// ---------------------------------------------------------------------------
// Kernel 1: fused QKV projection GEMM.
// Block = 64 rows (one window) x 64 out-cols, 128 threads (4 waves).
// Writes q (scaled, +bias) [b,h,n,d], k [b,h,n,d], v transposed [b,h,d,n].
// ---------------------------------------------------------------------------
__global__ void __launch_bounds__(128)
qkv_gemm(const float* __restrict__ x,
         const unsigned short* __restrict__ wqkv,
         const float* __restrict__ bq,
         const float* __restrict__ bkv,
         unsigned short* __restrict__ qw,
         unsigned short* __restrict__ kw,
         unsigned short* __restrict__ vtw) {
    __shared__ unsigned short As[64][32];   // x chunk, bf16, row = token
    __shared__ unsigned short Bs[64][32];   // W rows n0..n0+63, k chunk

    const int b    = blockIdx.x;            // window index (64 tokens)
    const int n0   = blockIdx.y * 64;       // output-column base (0..704)
    const int tid  = threadIdx.x;
    const int lane = tid & 31;
    const int wv   = tid >> 5;              // wave 0..3 -> 16-row strip
    const int l15  = lane & 15;
    const int hi   = lane >> 4;

    const float* xb = x + (size_t)b * NTOK * DIMC;

    v8f acc[4];
#pragma unroll
    for (int i = 0; i < 4; ++i) acc[i] = vzero8();

    for (int k0 = 0; k0 < DIMC; k0 += 32) {
        __syncthreads();
        // B tile: async global(bf16) -> LDS, 16B per issue
#pragma unroll
        for (int t = tid; t < 256; t += 128) {
            const int r = t >> 2, c = (t & 3) * 8;
            async_copy_b128(&wqkv[(size_t)(n0 + r) * DIMC + k0 + c], &Bs[r][c]);
        }
        // A tile: f32 -> bf16 convert-stage (needs VALU anyway)
        for (int t = tid; t < 64 * 32; t += 128) {
            const int r = t >> 5, c = t & 31;
            As[r][c] = f2bf(xb[(size_t)r * DIMC + k0 + c]);
        }
        // prefetch next x chunk (global_prefetch_b8)
        if (k0 + 32 < DIMC)
            __builtin_prefetch(&xb[(size_t)(tid >> 1) * DIMC + k0 + 32], 0, 0);
        wait_async();
        __syncthreads();

        // A fragment: lanes 0-15 -> K {0..7,16..23}; lanes 16-31 -> K {8..15,24..31}
        FragB16 af;
        const int am = wv * 16 + l15;
        const int kb = hi ? 8 : 0;
        af.u[0] = *(const u32x4*)&As[am][kb];
        af.u[1] = *(const u32x4*)&As[am][16 + kb];

#pragma unroll
        for (int ni = 0; ni < 4; ++ni) {
            // B fragment: lanes 0-15 -> K {0..7,16..23}; lanes 16-31 -> K {8..15,24..31}
            FragB16 bf_;
            const int bn = ni * 16 + l15;
            const int ko = hi ? 8 : 0;
            bf_.u[0] = *(const u32x4*)&Bs[bn][ko];
            bf_.u[1] = *(const u32x4*)&Bs[bn][ko + 16];
            acc[ni] = WMMA_BF16(af.v, bf_.v, acc[ni]);
        }
    }

    // Scatter-store into q / k / vT with bias (+scale on q)
#pragma unroll
    for (int ni = 0; ni < 4; ++ni) {
#pragma unroll
        for (int r = 0; r < 8; ++r) {
            const int m = wv * 16 + r + 8 * hi;     // token within window
            const int c = n0 + ni * 16 + l15;       // global out col [0,768)
            const float vacc = acc[ni][r];
            if (c < 256) {
                const float qv = (vacc + bq[c]) * SCALE_Q;
                const int h = c >> 5, d = c & 31;
                qw[(((size_t)b * HEADS + h) * NTOK + m) * HDIM + d] = f2bf(qv);
            } else if (c < 512) {
                const int c2 = c - 256;
                const float kv = vacc + bkv[c2];
                const int h = c2 >> 5, d = c2 & 31;
                kw[(((size_t)b * HEADS + h) * NTOK + m) * HDIM + d] = f2bf(kv);
            } else {
                const int c3 = c - 512;
                const float vv = vacc + bkv[c3 + 256];
                const int h = c3 >> 5, d = c3 & 31;
                vtw[(((size_t)b * HEADS + h) * HDIM + d) * NTOK + m] = f2bf(vv);
            }
        }
    }
}

// ---------------------------------------------------------------------------
// Kernel 2: per-(window, head) attention.
// 128 threads (4 waves); wave w owns token rows [16w, 16w+16).
// K and V^T tiles async-staged to LDS once; S = Q*K^T (+bias+mask) ->
// LDS softmax -> O = P*V -> bf16 [b, n, h*32+d].
// ---------------------------------------------------------------------------
__global__ void __launch_bounds__(128)
win_attn(const unsigned short* __restrict__ qw,
         const unsigned short* __restrict__ kw,
         const unsigned short* __restrict__ vtw,
         const float* __restrict__ bias_table,
         const int* __restrict__ rp_index,
         const float* __restrict__ mask,
         unsigned short* __restrict__ ao) {
    __shared__ float          S[64][64];     // scores f32
    __shared__ unsigned short P[64][64];     // probs bf16
    __shared__ unsigned short Ks[64][32];    // K tile  [n][d]
    __shared__ unsigned short Vs[32][64];    // V^T tile [d][n]

    const int b    = blockIdx.x;
    const int h    = blockIdx.y;
    const int tid  = threadIdx.x;
    const int lane = tid & 31;
    const int wv   = tid >> 5;
    const int l15  = lane & 15;
    const int hi   = lane >> 4;

    const unsigned short* qg = qw  + ((size_t)b * HEADS + h) * NTOK * HDIM; // [n][d]
    const unsigned short* kg = kw  + ((size_t)b * HEADS + h) * NTOK * HDIM; // [n][d]
    const unsigned short* vg = vtw + ((size_t)b * HEADS + h) * HDIM * NTOK; // [d][n]

    // --- async stage K (64x32) and V^T (32x64) tiles into LDS ---
#pragma unroll
    for (int t = tid; t < 256; t += 128) {
        const int r = t >> 2, c = (t & 3) * 8;
        async_copy_b128(&kg[r * HDIM + c], &Ks[r][c]);
    }
#pragma unroll
    for (int t = tid; t < 256; t += 128) {
        const int r = t >> 3, c = (t & 7) * 8;
        async_copy_b128(&vg[r * NTOK + c], &Vs[r][c]);
    }

    // Q fragment straight from global (each row read exactly once)
    FragB16 qf;
    {
        const int m  = wv * 16 + l15;
        const int kb = hi ? 8 : 0;
        qf.u[0] = *(const u32x4*)&qg[m * HDIM + kb];
        qf.u[1] = *(const u32x4*)&qg[m * HDIM + 16 + kb];
    }
    wait_async();
    __syncthreads();

    // --- S = Q * K^T : one 16x16x32 WMMA per 16x16 tile (K-depth == HDIM) ---
    v8f sacc[4];
#pragma unroll
    for (int i = 0; i < 4; ++i) sacc[i] = vzero8();
#pragma unroll
    for (int ni = 0; ni < 4; ++ni) {
        FragB16 kf;                           // B[k][n] = K[n][k]: rows of K
        const int n  = ni * 16 + l15;
        const int ko = hi ? 8 : 0;
        kf.u[0] = *(const u32x4*)&Ks[n][ko];
        kf.u[1] = *(const u32x4*)&Ks[n][ko + 16];
        sacc[ni] = WMMA_BF16(qf.v, kf.v, sacc[ni]);
    }

    // --- add relative-position bias + window mask, spill to LDS ---
    const int widx = b & (NW - 1);
#pragma unroll
    for (int ni = 0; ni < 4; ++ni) {
#pragma unroll
        for (int r = 0; r < 8; ++r) {
            const int m = wv * 16 + r + 8 * hi;
            const int n = ni * 16 + l15;
            const int idx = rp_index[m * NTOK + n];
            float sv = sacc[ni][r]
                     + bias_table[idx * HEADS + h]
                     + mask[(size_t)widx * (NTOK * NTOK) + m * NTOK + n];
            S[m][n] = sv;
        }
    }
    __syncthreads();

    // --- row softmax: one thread per token row ---
    if (tid < NTOK) {
        float mx = -3.0e38f;
        for (int n = 0; n < NTOK; ++n) mx = fmaxf(mx, S[tid][n]);
        float sum = 0.0f;
        for (int n = 0; n < NTOK; ++n) {
            float e = __expf(S[tid][n] - mx);
            S[tid][n] = e;
            sum += e;
        }
        const float rs = 1.0f / sum;
        for (int n = 0; n < NTOK; ++n) P[tid][n] = f2bf(S[tid][n] * rs);
    }
    __syncthreads();

    // --- O = P * V : K = 64 -> 2 chunks of 32, 2 d-tiles of 16 ---
#pragma unroll
    for (int dt = 0; dt < 2; ++dt) {
        v8f oacc = vzero8();
#pragma unroll
        for (int kc = 0; kc < 2; ++kc) {
            FragB16 pf;
            const int m  = wv * 16 + l15;
            const int kb = kc * 32 + (hi ? 8 : 0);
            pf.u[0] = *(const u32x4*)&P[m][kb];
            pf.u[1] = *(const u32x4*)&P[m][kb + 16];

            FragB16 vf;                        // B[k][d] = V[k][d]; Vs is [d][k]
            const int n  = dt * 16 + l15;
            const int ko = kc * 32 + (hi ? 8 : 0);
            vf.u[0] = *(const u32x4*)&Vs[n][ko];
            vf.u[1] = *(const u32x4*)&Vs[n][ko + 16];
            oacc = WMMA_BF16(pf.v, vf.v, oacc);
        }
#pragma unroll
        for (int r = 0; r < 8; ++r) {
            const int m = wv * 16 + r + 8 * hi;
            const int d = dt * 16 + l15;
            ao[((size_t)b * NTOK + m) * DIMC + h * HDIM + d] = f2bf(oacc[r]);
        }
    }
}

// ---------------------------------------------------------------------------
// Kernel 3: output projection GEMM (+b_o), fp32 result.
// Both operands are bf16 in memory -> pure async LDS staging.
// ---------------------------------------------------------------------------
__global__ void __launch_bounds__(128)
out_gemm(const unsigned short* __restrict__ ao,
         const unsigned short* __restrict__ wo,
         const float* __restrict__ bo,
         float* __restrict__ out) {
    __shared__ unsigned short As[64][32];
    __shared__ unsigned short Bs[64][32];

    const size_t row0 = (size_t)blockIdx.x * 64;
    const int n0   = blockIdx.y * 64;
    const int tid  = threadIdx.x;
    const int lane = tid & 31;
    const int wv   = tid >> 5;
    const int l15  = lane & 15;
    const int hi   = lane >> 4;

    v8f acc[4];
#pragma unroll
    for (int i = 0; i < 4; ++i) acc[i] = vzero8();

    for (int k0 = 0; k0 < DIMC; k0 += 32) {
        __syncthreads();
#pragma unroll
        for (int t = tid; t < 256; t += 128) {
            const int r = t >> 2, c = (t & 3) * 8;
            async_copy_b128(&ao[(row0 + r) * DIMC + k0 + c], &As[r][c]);
        }
#pragma unroll
        for (int t = tid; t < 256; t += 128) {
            const int r = t >> 2, c = (t & 3) * 8;
            async_copy_b128(&wo[(size_t)(n0 + r) * DIMC + k0 + c], &Bs[r][c]);
        }
        wait_async();
        __syncthreads();

        FragB16 af;
        const int am = wv * 16 + l15;
        const int kb = hi ? 8 : 0;
        af.u[0] = *(const u32x4*)&As[am][kb];
        af.u[1] = *(const u32x4*)&As[am][16 + kb];

#pragma unroll
        for (int ni = 0; ni < 4; ++ni) {
            FragB16 bf_;
            const int bn = ni * 16 + l15;
            const int ko = hi ? 8 : 0;
            bf_.u[0] = *(const u32x4*)&Bs[bn][ko];
            bf_.u[1] = *(const u32x4*)&Bs[bn][ko + 16];
            acc[ni] = WMMA_BF16(af.v, bf_.v, acc[ni]);
        }
    }

#pragma unroll
    for (int ni = 0; ni < 4; ++ni) {
#pragma unroll
        for (int r = 0; r < 8; ++r) {
            const int m = wv * 16 + r + 8 * hi;
            const int c = n0 + ni * 16 + l15;
            out[(row0 + m) * DIMC + c] = acc[ni][r] + bo[c];
        }
    }
}

// ---------------------------------------------------------------------------
// Launch
// ---------------------------------------------------------------------------
extern "C" void kernel_launch(void* const* d_in, const int* in_sizes, int n_in,
                              void* d_out, int out_size, void* d_ws, size_t ws_size,
                              hipStream_t stream) {
    const float* x          = (const float*)d_in[0];
    const float* mask       = (const float*)d_in[1];
    const float* w_q        = (const float*)d_in[2];
    const float* b_q        = (const float*)d_in[3];
    const float* w_kv       = (const float*)d_in[4];
    const float* b_kv       = (const float*)d_in[5];
    const float* bias_table = (const float*)d_in[6];
    const float* w_o        = (const float*)d_in[7];
    const float* b_o        = (const float*)d_in[8];
    const int*   rp_index   = (const int*)d_in[9];
    float* out = (float*)d_out;

    // Workspace layout (bytes)
    char* ws = (char*)d_ws;
    const size_t QKV_ELEMS = (size_t)BWIN * HEADS * NTOK * HDIM;   // 67,108,864
    unsigned short* wqkv = (unsigned short*)(ws);                                  // 768*256*2
    unsigned short* wo   = (unsigned short*)(ws + 393216);                         // 256*256*2
    unsigned short* qw   = (unsigned short*)(ws + 524288);
    unsigned short* kw   = (unsigned short*)(ws + 524288 + 2 * QKV_ELEMS);
    unsigned short* vtw  = (unsigned short*)(ws + 524288 + 4 * QKV_ELEMS);
    unsigned short* aow  = (unsigned short*)(ws + 524288 + 6 * QKV_ELEMS);

    cvt_weights<<<1024, 256, 0, stream>>>(w_q, w_kv, w_o, wqkv, wo);
    qkv_gemm<<<dim3(BWIN, 12), 128, 0, stream>>>(x, wqkv, b_q, b_kv, qw, kw, vtw);
    win_attn<<<dim3(BWIN, HEADS), 128, 0, stream>>>(qw, kw, vtw, bias_table, rp_index, mask, aow);
    out_gemm<<<dim3(BWIN, 4), 128, 0, stream>>>(aow, wo, b_o, out);
}
